// MultiHeadedAttention_23356032156262
// MI455X (gfx1250) — matrix operand order repaired and checked by reading, not compiled, not run on hardware
//
#include <hip/hip_runtime.h>
#include <hip/hip_bf16.h>
#include <math.h>
#include <stdint.h>

// ---------------------------------------------------------------------------
// MI455X (gfx1250) multi-head attention forward.
// B=2, S=2048, E=1024, H=16, Dh=64.  ~69 GFLOP vs ~100MB HBM traffic ->
// compute-bound: all matmuls via v_wmma_f32_16x16x32_f16 (wave32 WMMA),
// K/V tiles staged into LDS via the Tensor Data Mover where available.
// ---------------------------------------------------------------------------

typedef __attribute__((ext_vector_type(16))) _Float16 v16h;
typedef __attribute__((ext_vector_type(8)))  _Float16 v8h;
typedef __attribute__((ext_vector_type(8)))  float    v8f;

union V16U { v16h v; v8h h[2]; };

#define WMMA_F16(a, b, c) \
  __builtin_amdgcn_wmma_f32_16x16x32_f16(false, (a), false, (b), (short)0, (c), false, false)

#if defined(__AMDGCN__) && __has_builtin(__builtin_amdgcn_tensor_load_to_lds)
#define USE_TDM 1
typedef unsigned int v4u_ __attribute__((ext_vector_type(4)));
typedef int          v8i_ __attribute__((ext_vector_type(8)));
typedef int          v4i_ __attribute__((ext_vector_type(4)));

// Issue one TDM 2D tile load (f16 elements) per the CDNA5 D# layout.
//   tensor: tensor_d0 x tensor_d1 elements, row stride stride_d0 (elements)
//   tile:   tile_d0 x tile_d1 elements starting at gaddr -> LDS lds_off
// This toolchain exposes the 6-arg builtin:
//   (uint32x4 g0, int32x8 g1, int32x4 g2, int32x4 g3, int32x8 extra, i32 cpol)
__device__ __forceinline__ void tdm_load_2d_f16(uint32_t lds_off, const void* gaddr,
                                                uint32_t tensor_d0, uint32_t tensor_d1,
                                                uint32_t tile_d0, uint32_t tile_d1,
                                                uint32_t stride_d0) {
  uint64_t ga = (uint64_t)(uintptr_t)gaddr;
  v4u_ g0;
  g0[0] = 1u;                                  // count=1, is_restore=0, gather off
  g0[1] = lds_off;                             // D#.lds_addr (bytes)
  g0[2] = (uint32_t)ga;                        // global_addr[31:0]   (bits 95:64)
  g0[3] = (uint32_t)(ga >> 32) | (2u << 30);   // global_addr[56:32], type=2 (image)
  v8i_ g1;
  g1[0] = (int)(1u << 16);                     // wg_mask=0, data_size=1 (2 bytes)
  g1[1] = (int)((tensor_d0 & 0xFFFFu) << 16);                      // dim0[15:0] @63:48
  g1[2] = (int)((tensor_d0 >> 16) | ((tensor_d1 & 0xFFFFu) << 16));// dim0 hi, dim1 lo
  g1[3] = (int)((tensor_d1 >> 16) | ((tile_d0 & 0xFFFFu) << 16));  // dim1 hi, tile_dim0
  g1[4] = (int)(tile_d1 & 0xFFFFu);                                // tile_dim1, tile_dim2=0
  g1[5] = (int)stride_d0;                                          // dim0_stride[31:0]
  g1[6] = 0;                                                       // dim0_stride hi, dim1_stride lo
  g1[7] = 0;
  v4i_ z4 = {0, 0, 0, 0};
  v8i_ z8 = {0, 0, 0, 0, 0, 0, 0, 0};
  __builtin_amdgcn_tensor_load_to_lds(g0, g1, z4, z4, z8, 0);
}
#else
#define USE_TDM 0
#endif

// ------------------------------ conversions --------------------------------

__global__ void k_f32_to_f16(const float* __restrict__ in, _Float16* __restrict__ out, int n) {
  int i = blockIdx.x * blockDim.x + threadIdx.x;
  if (i < n) out[i] = (_Float16)in[i];
}

// in[R][C] (f32) -> out[C][R] (f16): store W^T so GEMM B-fragments are contiguous
__global__ void k_transpose_f32_to_f16(const float* __restrict__ in, _Float16* __restrict__ out,
                                       int R, int C) {
  int i = blockIdx.x * blockDim.x + threadIdx.x;
  if (i < R * C) {
    int r = i / C, c = i % C;
    out[(size_t)c * R + r] = (_Float16)in[i];
  }
}

// ------------------------------- WMMA GEMM ---------------------------------
// C[M,N] = A[M,K] * Bt[N,K]^T + bias[N]
// Each wave computes a 16x64 output tile: one A-fragment reused across 4
// B-fragments / 4 WMMAs per K-step (cuts load-per-WMMA and A L2 traffic 4x).
// mode 0: f16 out as [B,H,S,Dh]   (Q, K projections)
// mode 1: f16 out as [B,H,Dh,S]   (V projection, pre-transposed for PV GEMM)
// mode 2: f32 out as [M,N]        (final output projection)
__global__ void k_gemm_wmma(const _Float16* __restrict__ A, const _Float16* __restrict__ Bt,
                            const float* __restrict__ bias, void* __restrict__ outp,
                            int M, int N, int K, int mode, int S_, int Dh_) {
  const int l    = threadIdx.x & 31;
  const int wave = threadIdx.x >> 5;
  const int lo   = l & 15;
  const int hi   = l >> 4;

  const int tile = blockIdx.x * (blockDim.x >> 5) + wave;
  const int ntn  = N >> 6;                 // 64-wide column tiles
  const int tm   = tile / ntn;
  const int tn   = tile % ntn;
  if (tm >= (M >> 4)) return;              // uniform per wave

  // A-fragment (16x32 f16): lane = row m=lo,
  //   K = 16*hi..+15 (contiguous, matches B).
  const _Float16* arow = A  + (size_t)(tm * 16 + lo) * K;
  // B-fragment (32x16 f16): lane = col n=lo, K = 16*hi..+15 (contiguous in Bt).
  const _Float16* brow = Bt + (size_t)(tn * 64 + lo) * K;

  v8f acc[4] = {};
  for (int k0 = 0; k0 < K; k0 += 32) {
    V16U af;
    af.h[0] = *(const v8h*)(arow + k0 + 16 * hi);
    af.h[1] = *(const v8h*)(arow + k0 + 16 * hi + 8);
    if (k0 + 32 < K) __builtin_prefetch((const void*)(arow + k0 + 32), 0, 1);
    for (int j = 0; j < 4; ++j) {
      v16h bf = *(const v16h*)(brow + (size_t)j * 16 * K + k0 + 16 * hi);
      if (k0 + 32 < K)
        __builtin_prefetch((const void*)(brow + (size_t)j * 16 * K + k0 + 32), 0, 1);
      acc[j] = WMMA_F16(af.v, bf, acc[j]);
    }
  }

  // D-fragment layout: lane holds N = lo, M = 8*hi + r.
  for (int j = 0; j < 4; ++j) {
    const int ncol = tn * 64 + j * 16 + lo;
    const float bv = bias ? bias[ncol] : 0.0f;
    if (mode == 2) {
      float* out = (float*)outp;
      for (int r = 0; r < 8; ++r) {
        int m = tm * 16 + 8 * hi + r;
        out[(size_t)m * N + ncol] = acc[j][r] + bv;
      }
    } else {
      _Float16* out = (_Float16*)outp;
      const int H_ = N / Dh_;
      const int h  = ncol / Dh_;
      const int d  = ncol % Dh_;
      for (int r = 0; r < 8; ++r) {
        int m = tm * 16 + 8 * hi + r;      // global row = b*S + s
        int b = m / S_, s = m % S_;
        size_t idx;
        if (mode == 0) idx = (((size_t)b * H_ + h) * S_ + s) * Dh_ + d;   // [B,H,S,Dh]
        else           idx = (((size_t)b * H_ + h) * Dh_ + d) * S_ + s;   // [B,H,Dh,S]
        out[idx] = (_Float16)(acc[j][r] + bv);
      }
    }
  }
}

// ---------------------------- flash attention ------------------------------
// 4 waves/block, each wave owns one 16-row q tile of one (b,h) and a private
// LDS slice. Streams kv in blocks of 32 with online softmax; causal mask is
// analytic. K/V tiles staged LDS via TDM when available.
// Q,K: [B,H,S,64] f16.  Vt: [B,H,64,S] f16.  Z out: [B,S,H*64] f16.
__global__ __launch_bounds__(128)
void k_flash_attn(const _Float16* __restrict__ Q, const _Float16* __restrict__ Kk,
                  const _Float16* __restrict__ Vt, _Float16* __restrict__ Z,
                  int S_, int H_) {
  const int Dh = 64;
  __shared__ __align__(32) _Float16 smemP[4][16 * 32];   // P: D-layout -> A-layout
#if USE_TDM
  __shared__ __align__(32) _Float16 smemK[4][32 * 64];   // K tile  [kv][d]
  __shared__ __align__(32) _Float16 smemV[4][64 * 32];   // Vt tile [d][kv]
#endif

  const int l    = threadIdx.x & 31;
  const int wave = threadIdx.x >> 5;
  const int lo   = l & 15;
  const int hi   = l >> 4;

  const int qtiles = S_ >> 4;
  const int tile = blockIdx.x * (blockDim.x >> 5) + wave;
  const int bh = tile / qtiles;
  const int q0 = (tile % qtiles) << 4;

  const _Float16* Qb = Q  + (size_t)bh * S_ * Dh;
  const _Float16* Kb = Kk + (size_t)bh * S_ * Dh;
  const _Float16* Vb = Vt + (size_t)bh * Dh * S_;

  // Q A-fragments for the two Dh K-steps (d0 = 0, 32); rows q0..q0+15.
  V16U aq[2];
  const _Float16* qrow = Qb + (size_t)(q0 + lo) * Dh;
  for (int t = 0; t < 2; ++t) {
    aq[t].h[0] = *(const v8h*)(qrow + 32 * t + 16 * hi);
    aq[t].h[1] = *(const v8h*)(qrow + 32 * t + 16 * hi + 8);
  }

  float rmax[8], rsum[8];
  v8f o[4] = {};
  for (int r = 0; r < 8; ++r) { rmax[r] = -1e30f; rsum[r] = 0.0f; }

  for (int kk0 = 0; kk0 <= q0; kk0 += 32) {
#if USE_TDM
    // Retire our previous LDS reads before the TDM overwrites the tiles
    // (TDM is unordered with DS ops), then DMA this step's K and V tiles.
    asm volatile("s_wait_dscnt 0" ::: "memory");
    tdm_load_2d_f16((uint32_t)(uintptr_t)&smemK[wave][0], Kb + (size_t)kk0 * Dh,
                    /*tensor_d0=*/Dh, /*tensor_d1=*/(uint32_t)(S_ - kk0),
                    /*tile_d0=*/Dh, /*tile_d1=*/32, /*stride_d0=*/Dh);
    tdm_load_2d_f16((uint32_t)(uintptr_t)&smemV[wave][0], Vb + kk0,
                    /*tensor_d0=*/(uint32_t)(S_ - kk0), /*tensor_d1=*/Dh,
                    /*tile_d0=*/32, /*tile_d1=*/Dh, /*stride_d0=*/(uint32_t)S_);
    __builtin_amdgcn_s_wait_tensorcnt(0);
#endif

    // ---- scores: two 16x16 tiles (kv columns kk0.. and kk0+16..) ----------
    v8f s0 = {}, s1 = {};
#if USE_TDM
    {
      const _Float16* kt = &smemK[wave][0];
      for (int t = 0; t < 2; ++t) {
        v16h b0 = *(const v16h*)(kt + (size_t)lo * Dh        + 32 * t + 16 * hi);
        v16h b1 = *(const v16h*)(kt + (size_t)(16 + lo) * Dh + 32 * t + 16 * hi);
        s0 = WMMA_F16(aq[t].v, b0, s0);
        s1 = WMMA_F16(aq[t].v, b1, s1);
      }
    }
#else
    {
      int n0 = kk0 + lo;
      int n1 = kk0 + 16 + lo; if (n1 > S_ - 1) n1 = S_ - 1;   // masked anyway
      const _Float16* k0p = Kb + (size_t)n0 * Dh + 16 * hi;
      const _Float16* k1p = Kb + (size_t)n1 * Dh + 16 * hi;
      for (int t = 0; t < 2; ++t) {
        v16h b0 = *(const v16h*)(k0p + 32 * t);
        v16h b1 = *(const v16h*)(k1p + 32 * t);
        s0 = WMMA_F16(aq[t].v, b0, s0);
        s1 = WMMA_F16(aq[t].v, b1, s1);
      }
    }
#endif

    // ---- online softmax (row = q0 + 8*hi + r, col kv = kk0(+16) + lo) -----
    float p0[8], p1[8];
    for (int r = 0; r < 8; ++r) {
      const int row = q0 + 8 * hi + r;
      const int kv0 = kk0 + lo;
      const int kv1 = kk0 + 16 + lo;
      float a0 = (kv0 <= row) ? s0[r] * 0.125f : -1e30f;   // 1/sqrt(64)
      float a1 = (kv1 <= row) ? s1[r] * 0.125f : -1e30f;
      float m = fmaxf(a0, a1);
      for (int msk = 1; msk < 16; msk <<= 1) m = fmaxf(m, __shfl_xor(m, msk, 32));
      float mnew = fmaxf(rmax[r], m);
      float e0 = __expf(a0 - mnew);
      float e1 = __expf(a1 - mnew);
      float ps = e0 + e1;
      for (int msk = 1; msk < 16; msk <<= 1) ps += __shfl_xor(ps, msk, 32);
      float corr = __expf(rmax[r] - mnew);
      rsum[r] = rsum[r] * corr + ps;
      rmax[r] = mnew;
      o[0][r] *= corr; o[1][r] *= corr; o[2][r] *= corr; o[3][r] *= corr;
      p0[r] = e0; p1[r] = e1;
    }

    // ---- P: D-layout -> LDS -> A-layout -----------------------------------
    for (int r = 0; r < 8; ++r) {
      smemP[wave][(8 * hi + r) * 32 + lo]      = (_Float16)p0[r];
      smemP[wave][(8 * hi + r) * 32 + 16 + lo] = (_Float16)p1[r];
    }
    asm volatile("s_wait_dscnt 0" ::: "memory");
    V16U ap;
    ap.h[0] = *(const v8h*)(&smemP[wave][lo * 32 + 16 * hi]);
    ap.h[1] = *(const v8h*)(&smemP[wave][lo * 32 + 16 * hi + 8]);

    // ---- PV: 4 WMMAs (one per 16-wide Dh slice) ---------------------------
#if USE_TDM
    for (int j = 0; j < 4; ++j) {
      v16h bv = *(const v16h*)(&smemV[wave][(16 * j + lo) * 32 + 16 * hi]);
      o[j] = WMMA_F16(ap.v, bv, o[j]);
    }
#else
    int kbase = kk0 + 16 * hi;
    if (kbase + 16 > S_) kbase = S_ - 16;   // OOB rows hit only zeroed P cols
    for (int j = 0; j < 4; ++j) {
      v16h bv = *(const v16h*)(Vb + (size_t)(16 * j + lo) * S_ + kbase);
      o[j] = WMMA_F16(ap.v, bv, o[j]);
    }
#endif
  }

  // ---- epilogue: normalize and scatter to Z[B,S,H*Dh] ---------------------
  const int b = bh / H_;
  const int h = bh % H_;
  const int HD = H_ * Dh;
  for (int r = 0; r < 8; ++r) {
    float inv = 1.0f / rsum[r];
    int row = q0 + 8 * hi + r;
    _Float16* zr = Z + ((size_t)b * S_ + row) * HD + h * Dh;
    for (int j = 0; j < 4; ++j) zr[16 * j + lo] = (_Float16)(o[j][r] * inv);
  }
}

// --------------------------------- launch ----------------------------------

extern "C" void kernel_launch(void* const* d_in, const int* in_sizes, int n_in,
                              void* d_out, int out_size, void* d_ws, size_t ws_size,
                              hipStream_t stream) {
  (void)in_sizes; (void)n_in; (void)out_size; (void)ws_size;
  const int B_ = 2, S_ = 2048, E_ = 1024, H_ = 16, Dh_ = 64;
  const int M_  = B_ * S_;        // 4096
  const int HD_ = H_ * Dh_;       // 1024

  const float* q_in = (const float*)d_in[0];
  const float* k_in = (const float*)d_in[1];
  const float* v_in = (const float*)d_in[2];
  // d_in[3] = causal mask: computed analytically, never read
  const float* WQ = (const float*)d_in[4];
  const float* bQ = (const float*)d_in[5];
  const float* WK = (const float*)d_in[6];
  const float* bK = (const float*)d_in[7];
  const float* WV = (const float*)d_in[8];
  const float* bV = (const float*)d_in[9];
  const float* WO = (const float*)d_in[10];
  const float* bO = (const float*)d_in[11];

  size_t off = 0;
  auto alloc = [&](size_t bytes) -> void* {
    void* p = (char*)d_ws + off;
    off += (bytes + 255) & ~(size_t)255;
    return p;
  };
  _Float16* Xq16 = (_Float16*)alloc((size_t)M_ * E_ * 2);
  _Float16* Xk16 = (_Float16*)alloc((size_t)M_ * E_ * 2);
  _Float16* Xv16 = (_Float16*)alloc((size_t)M_ * E_ * 2);
  _Float16* WQt  = (_Float16*)alloc((size_t)E_ * HD_ * 2);
  _Float16* WKt  = (_Float16*)alloc((size_t)E_ * HD_ * 2);
  _Float16* WVt  = (_Float16*)alloc((size_t)E_ * HD_ * 2);
  _Float16* WOt  = (_Float16*)alloc((size_t)HD_ * E_ * 2);
  _Float16* Qh   = (_Float16*)alloc((size_t)M_ * HD_ * 2);   // [B,H,S,Dh]
  _Float16* Kh   = (_Float16*)alloc((size_t)M_ * HD_ * 2);   // [B,H,S,Dh]
  _Float16* Vth  = (_Float16*)alloc((size_t)M_ * HD_ * 2);   // [B,H,Dh,S]
  _Float16* Zh   = (_Float16*)alloc((size_t)M_ * HD_ * 2);   // [B,S,H*Dh]

  const int nElem = M_ * E_;
  dim3 cgrid((nElem + 255) / 256), cblk(256);
  k_f32_to_f16<<<cgrid, cblk, 0, stream>>>(q_in, Xq16, nElem);
  k_f32_to_f16<<<cgrid, cblk, 0, stream>>>(k_in, Xk16, nElem);
  k_f32_to_f16<<<cgrid, cblk, 0, stream>>>(v_in, Xv16, nElem);

  const int nW = E_ * HD_;
  dim3 tgrid((nW + 255) / 256);
  k_transpose_f32_to_f16<<<tgrid, cblk, 0, stream>>>(WQ, WQt, E_, HD_);
  k_transpose_f32_to_f16<<<tgrid, cblk, 0, stream>>>(WK, WKt, E_, HD_);
  k_transpose_f32_to_f16<<<tgrid, cblk, 0, stream>>>(WV, WVt, E_, HD_);
  k_transpose_f32_to_f16<<<tgrid, cblk, 0, stream>>>(WO, WOt, HD_, E_);

  // projections: (4096/16)*(1024/64) = 4096 wave-tiles, 4 waves/block
  const int tiles = (M_ / 16) * (HD_ / 64);
  dim3 ggrid(tiles / 4), gblk(128);
  k_gemm_wmma<<<ggrid, gblk, 0, stream>>>(Xq16, WQt, bQ, Qh,  M_, HD_, E_, 0, S_, Dh_);
  k_gemm_wmma<<<ggrid, gblk, 0, stream>>>(Xk16, WKt, bK, Kh,  M_, HD_, E_, 0, S_, Dh_);
  k_gemm_wmma<<<ggrid, gblk, 0, stream>>>(Xv16, WVt, bV, Vth, M_, HD_, E_, 1, S_, Dh_);

  // flash attention: 2*16*128 = 4096 q-tiles, 4 waves/block -> 1024 blocks
  dim3 fgrid(B_ * H_ * (S_ / 16) / 4), fblk(128);
  k_flash_attn<<<fgrid, fblk, 0, stream>>>(Qh, Kh, Vth, Zh, S_, H_);

  // output projection -> f32 d_out
  k_gemm_wmma<<<ggrid, gblk, 0, stream>>>(Zh, WOt, bO, d_out, M_, E_, HD_, 2, S_, Dh_);
}
